// NNUE_43525198577726
// MI455X (gfx1250) — compile-verified, matrix operand order checked by reading, not run on hardware
//
#include <hip/hip_runtime.h>
#include <hip/hip_bf16.h>
#include <stdint.h>

#define NF   20480
#define NACC 128
#define NB   4096
#define NWT  144            // padded weight rows: 128 acc + 2 psqt + 14 zero

#define KC     256          // K-chunk staged in LDS
#define KSTEPS (KC / 32)    // 8 WMMA K-steps per chunk
#define NCHUNK (NF / KC)    // 80
#define LROW   (KC + 8)     // padded row length (elements): 264
#define LROWB  (LROW * 2)   // 528 bytes = 132 dwords == 4 mod 64 banks
#define LDSBUF (NWT * LROWB)            // 76032 B per buffer
#define SEGS_PER_THREAD ((NWT * (KC * 2 / 16)) / 128)   // 4608/128 = 36

typedef __bf16 bf16_t;
typedef __attribute__((ext_vector_type(16))) __bf16 v16bf;
typedef __attribute__((ext_vector_type(8)))  float  v8f;

union V16 {
    unsigned short s[16];
    unsigned       d[8];
    uint4          u[2];
    v16bf          v;
};

union B8 {
    unsigned short s[8];
    uint4          u;
};

__device__ __forceinline__ unsigned short bf16_rne(float f) {
    unsigned int ui = __float_as_uint(f);
    ui += 0x7FFFu + ((ui >> 16) & 1u);
    return (unsigned short)(ui >> 16);
}

// pack two floats' bf16 truncations (exact for 0.0/1.0) into one dword:
// low half = lo, high half = hi  ->  single v_perm_b32
__device__ __forceinline__ unsigned pack_bf16_pair(float lo, float hi) {
    return __builtin_amdgcn_perm(__float_as_uint(hi), __float_as_uint(lo),
                                 0x07060302u);
}

// CDNA5 async global -> LDS copy, 16 B per lane, tracked by ASYNCcnt.
__device__ __forceinline__ void async_ld_b128(unsigned ldsaddr, const void* gaddr) {
    asm volatile("global_load_async_to_lds_b128 %0, %1, off"
                 :: "v"(ldsaddr), "v"(gaddr) : "memory");
}

// ---------------------------------------------------------------------------
// Pack acc_w (rows 0..127) + psqt_w (rows 128..129) + zeros (130..143) into
// bf16 row-major Wb[144][NF] in workspace.
// ---------------------------------------------------------------------------
__global__ void nnue_prep_weights(const float* __restrict__ acc_w,
                                  const float* __restrict__ psqt_w,
                                  bf16_t* __restrict__ Wb) {
    const int n  = blockIdx.y;
    const int kk = (blockIdx.x * blockDim.x + threadIdx.x) * 8;
    B8 pk;
    if (n < NACC) {
        const float* src = acc_w + (size_t)n * NF + kk;
        #pragma unroll
        for (int i = 0; i < 8; ++i) pk.s[i] = bf16_rne(src[i]);
    } else if (n < NACC + 2) {
        const float* src = psqt_w + (size_t)(n - NACC) * NF + kk;
        #pragma unroll
        for (int i = 0; i < 8; ++i) pk.s[i] = bf16_rne(src[i]);
    } else {
        #pragma unroll
        for (int i = 0; i < 8; ++i) pk.s[i] = 0;
    }
    *(uint4*)(Wb + (size_t)n * NF + kk) = pk.u;
}

__global__ void nnue_zero_out(float* __restrict__ out) {
    const int i = blockIdx.x * blockDim.x + threadIdx.x;
    if (i < NB * 2) out[i] = 0.0f;
}

// ---------------------------------------------------------------------------
// Fused NNUE kernel. Block = 128 threads (4 waves), each wave owns two 16-row
// M-tiles (32 rows); block covers 128 rows of one side. Weights are staged
// per K-chunk into a double-buffered LDS tile via async global->LDS copies.
// ---------------------------------------------------------------------------
__global__ void __launch_bounds__(128)
nnue_gemm(const float* __restrict__ white, const float* __restrict__ black,
          const bf16_t* __restrict__ Wb,
          const float* __restrict__ acc_b, const float* __restrict__ out_w,
          float* __restrict__ out) {
    __shared__ char SMEM[2 * LDSBUF];    // 152064 B (gfx1250 WGP: 320 KB LDS)

    const int tid  = threadIdx.x;
    const int lane = tid & 31;
    const int wave = tid >> 5;                       // 0..3
    const int side = blockIdx.y;
    const float sign = side ? -1.0f : 1.0f;
    const float* __restrict__ x = side ? black : white;

    const int M0 = blockIdx.x * 128 + wave * 32;     // first row of tile pair
    const int cl = lane & 15;                        // row (A) / col (B,C) in tile
    const int kg = (lane >> 4) * 8;                  // K-subgroup base (§7.12.2)

    const float* __restrict__ xr0 = x + (size_t)(M0 + cl) * NF;
    const float* __restrict__ xr1 = x + (size_t)(M0 + 16 + cl) * NF;

    const unsigned smembase = (unsigned)(uintptr_t)SMEM;  // LDS byte offset

    v8f C[18];
    #pragma unroll
    for (int t = 0; t < 18; ++t) C[t] = (v8f){0.f,0.f,0.f,0.f,0.f,0.f,0.f,0.f};

    // ---- stage chunk 0 into buffer 0
    {
        #pragma unroll
        for (int i = 0; i < SEGS_PER_THREAD; ++i) {
            const int idx = i * 128 + tid;           // 0..4607
            const int n   = idx >> 5;                // weight row
            const int seg = idx & 31;                // 16B segment in row
            async_ld_b128(smembase + (unsigned)n * LROWB + (unsigned)seg * 16,
                          (const void*)(Wb + (size_t)n * NF + seg * 8));
        }
    }

    for (int c = 0; c < NCHUNK; ++c) {
        const int kbase = c * KC;

        if (c + 1 < NCHUNK) {
            // issue next chunk into the other buffer (safe: last read 2 chunks ago)
            const unsigned woff = smembase + (unsigned)((c + 1) & 1) * LDSBUF;
            const int knext = kbase + KC;
            #pragma unroll
            for (int i = 0; i < SEGS_PER_THREAD; ++i) {
                const int idx = i * 128 + tid;
                const int n   = idx >> 5;
                const int seg = idx & 31;
                async_ld_b128(woff + (unsigned)n * LROWB + (unsigned)seg * 16,
                              (const void*)(Wb + (size_t)n * NF + knext + seg * 8));
            }
            // async copies complete in order: <=36 outstanding => chunk c done
            asm volatile("s_wait_asynccnt 0x24" ::: "memory");
        } else {
            asm volatile("s_wait_asynccnt 0x0" ::: "memory");
        }
        __syncthreads();                              // chunk c visible to all waves

        #pragma unroll 2
        for (int ks = 0; ks < KSTEPS; ++ks) {
            const int g0 = kbase + ks * 32 + kg;      // global K index
            const int gl = ks * 32 + kg;              // K index within chunk

            // A tiles: 16 floats/lane each (K g..g+7 and g+16..g+23), b128 loads
            const float4 a0 = *(const float4*)(xr0 + g0);
            const float4 a1 = *(const float4*)(xr0 + g0 + 4);
            const float4 a2 = *(const float4*)(xr0 + g0 + 16);
            const float4 a3 = *(const float4*)(xr0 + g0 + 20);
            const float4 b0 = *(const float4*)(xr1 + g0);
            const float4 b1 = *(const float4*)(xr1 + g0 + 4);
            const float4 b2 = *(const float4*)(xr1 + g0 + 16);
            const float4 b3 = *(const float4*)(xr1 + g0 + 20);
            __builtin_prefetch(xr0 + g0 + 512, 0, 1); // global_prefetch_b8
            __builtin_prefetch(xr1 + g0 + 512, 0, 1);

            V16 A0v, A1v;                              // v_perm_b32 packing
            A0v.d[0] = pack_bf16_pair(a0.x, a0.y);
            A0v.d[1] = pack_bf16_pair(a0.z, a0.w);
            A0v.d[2] = pack_bf16_pair(a1.x, a1.y);
            A0v.d[3] = pack_bf16_pair(a1.z, a1.w);
            A0v.d[4] = pack_bf16_pair(a2.x, a2.y);
            A0v.d[5] = pack_bf16_pair(a2.z, a2.w);
            A0v.d[6] = pack_bf16_pair(a3.x, a3.y);
            A0v.d[7] = pack_bf16_pair(a3.z, a3.w);
            A1v.d[0] = pack_bf16_pair(b0.x, b0.y);
            A1v.d[1] = pack_bf16_pair(b0.z, b0.w);
            A1v.d[2] = pack_bf16_pair(b1.x, b1.y);
            A1v.d[3] = pack_bf16_pair(b1.z, b1.w);
            A1v.d[4] = pack_bf16_pair(b2.x, b2.y);
            A1v.d[5] = pack_bf16_pair(b2.z, b2.w);
            A1v.d[6] = pack_bf16_pair(b3.x, b3.y);
            A1v.d[7] = pack_bf16_pair(b3.z, b3.w);

            #pragma unroll
            for (int t = 0; t < 9; ++t) {
                const char* bp = SMEM + (size_t)(c & 1) * LDSBUF
                               + (size_t)(t * 16 + cl) * LROWB + (size_t)gl * 2;
                V16 B;                                 // ds_load_b128 x2
                B.u[0] = *(const uint4*)(bp);
                B.u[1] = *(const uint4*)(bp + 32);
                C[t] = __builtin_amdgcn_wmma_f32_16x16x32_bf16(
                    false, A0v.v, false, B.v, (short)0, C[t], false, false);
                C[9 + t] = __builtin_amdgcn_wmma_f32_16x16x32_bf16(
                    false, A1v.v, false, B.v, (short)0, C[9 + t], false, false);
            }
        }
        __syncthreads();                              // done reading before reuse
    }

    // ---- epilogue: bias + CReLU + 128->2 layer + psqt, per tile group
    float bcol[8], w0[8], w1[8];
    #pragma unroll
    for (int t = 0; t < 8; ++t) {
        const int col = t * 16 + cl;
        bcol[t] = acc_b[col];
        w0[t]   = out_w[col];            // out_w[0][c]
        w1[t]   = out_w[NACC + col];     // out_w[1][c]
    }

    #pragma unroll
    for (int gidx = 0; gidx < 2; ++gidx) {
        float s0[8], s1[8];
        #pragma unroll
        for (int r = 0; r < 8; ++r) { s0[r] = 0.f; s1[r] = 0.f; }

        #pragma unroll
        for (int t = 0; t < 8; ++t) {
            #pragma unroll
            for (int r = 0; r < 8; ++r) {
                float a = C[gidx * 9 + t][r] + bcol[t];
                a = fminf(fmaxf(a, 0.0f), 1.0f);      // CReLU
                s0[r] += a * w0[t];
                s1[r] += a * w1[t];
            }
        }
        #pragma unroll
        for (int r = 0; r < 8; ++r) {                 // psqt cols 0/1 of tile 8
            if (cl == 0) s0[r] += C[gidx * 9 + 8][r];
            if (cl == 1) s1[r] += C[gidx * 9 + 8][r];
        }
        // reduce over the 16 lanes of each half (rows 0-7 lanes 0-15, 8-15 up)
        #pragma unroll
        for (int r = 0; r < 8; ++r) {
            #pragma unroll
            for (int off = 8; off >= 1; off >>= 1) {
                s0[r] += __shfl_xor(s0[r], off, 32);
                s1[r] += __shfl_xor(s1[r], off, 32);
            }
        }
        if (cl == 0) {
            const int mbase = M0 + gidx * 16 + (lane >> 4) * 8;
            #pragma unroll
            for (int r = 0; r < 8; ++r) {
                atomicAdd(&out[(mbase + r) * 2 + 0], sign * s0[r]);
                atomicAdd(&out[(mbase + r) * 2 + 1], sign * s1[r]);
            }
        }
    }
}

extern "C" void kernel_launch(void* const* d_in, const int* in_sizes, int n_in,
                              void* d_out, int out_size, void* d_ws, size_t ws_size,
                              hipStream_t stream) {
    const float* white  = (const float*)d_in[0];
    const float* black  = (const float*)d_in[1];
    const float* psqt_w = (const float*)d_in[2];
    const float* acc_w  = (const float*)d_in[3];
    const float* acc_b  = (const float*)d_in[4];
    const float* out_w  = (const float*)d_in[5];
    float*  out = (float*)d_out;
    bf16_t* Wb  = (bf16_t*)d_ws;    // needs NWT*NF*2 = 5.9 MB of workspace

    dim3 pgrid(NF / (256 * 8), NWT);                     // 10 x 144
    nnue_prep_weights<<<pgrid, 256, 0, stream>>>(acc_w, psqt_w, Wb);

    nnue_zero_out<<<(NB * 2 + 255) / 256, 256, 0, stream>>>(out);

    dim3 ggrid(NB / 128, 2);                             // 32 x 2 (side)
    nnue_gemm<<<ggrid, 128, 0, stream>>>(white, black, Wb, acc_b, out_w, out);
}
